// SparseMoeBlock_68719476736412
// MI455X (gfx1250) — compile-verified, matrix operand order checked by reading, not run on hardware
//
#include <hip/hip_runtime.h>
#include <math.h>
#include <stdint.h>

// ---------------------------------------------------------------- types
typedef __bf16 bf16_t;
typedef __attribute__((ext_vector_type(8)))  bf16_t v8bf;
typedef __attribute__((ext_vector_type(16))) bf16_t v16bf;
typedef __attribute__((ext_vector_type(8)))  float  v8f;
typedef int v4i __attribute__((vector_size(16)));   // matches builtin's V4i

// ---------------------------------------------------------------- sizes
#define DD   1024      // embed dim
#define EE   16        // experts
#define II   4096      // expert intermediate
#define ISH  2048      // shared intermediate
#define TT   4096      // tokens (B*S)
#define CC   512       // capacity per expert
#define LDSW 40        // padded bf16 row stride (80B) for 16B-aligned, conflict-free tiles

// ---------------------------------------------------------------- async path (guarded)
#if defined(__has_builtin)
#if __has_builtin(__builtin_amdgcn_global_load_async_to_lds_b128) && \
    __has_builtin(__builtin_amdgcn_s_wait_asynccnt)
#define HAVE_ASYNC 1
#endif
#endif
#ifndef HAVE_ASYNC
#define HAVE_ASYNC 0
#endif

#if HAVE_ASYNC
#define ASG __attribute__((address_space(1)))
#define ASL __attribute__((address_space(3)))
// One lane copies 16B global -> LDS, tracked by ASYNCcnt (no VGPR destination).
__device__ __forceinline__ void async_cp16(const bf16_t* g, bf16_t* l) {
  __builtin_amdgcn_global_load_async_to_lds_b128(
      (ASG v4i*)(uintptr_t)g,
      (ASL v4i*)(unsigned)(uintptr_t)l, 0, 0);
}
__device__ __forceinline__ void wait_async0() {
  __builtin_amdgcn_s_wait_asynccnt(0);
}
#endif

// ---------------------------------------------------------------- helpers
__device__ __forceinline__ v8f wmma_bf16(v16bf a, v16bf b, v8f c) {
  // D = A(16x32 bf16) * B(32x16 bf16) + C(16x16 f32)
  return __builtin_amdgcn_wmma_f32_16x16x32_bf16(false, a, false, b,
                                                 (short)0, c, false, false);
}

// Load one lane's 16 bf16 elements of a 16x32 A/B tile from a K-contiguous
// row (p already includes the +8*half element offset). Two 16B chunks:
// elements [0..7] -> VGPR0..3 (K = 8h..8h+7), [8..15] -> VGPR4..7 (K = 16+8h..).
__device__ __forceinline__ v16bf load_tile_bf16(const bf16_t* p) {
  v8bf lo = *(const v8bf*)(p);
  v8bf hi = *(const v8bf*)(p + 16);
  v16bf r;
#pragma unroll
  for (int i = 0; i < 8; ++i) { r[i] = lo[i]; r[i + 8] = hi[i]; }
  return r;
}

// Stage a 32(N) x 32(K) f32 weight tile into LDS as bf16 (row stride LDSW).
// 256 threads: each loads one float4 and converts.
__device__ __forceinline__ void stage_w32x32(const float* __restrict__ W, int ldw,
                                             int n0, int k0, bf16_t* lds, int tid) {
  const int row = tid >> 3;          // 0..31
  const int col = (tid & 7) * 4;     // 0..28
  const float4 v = *(const float4*)(W + (size_t)(n0 + row) * ldw + k0 + col);
  bf16_t* d = lds + row * LDSW + col;
  d[0] = (bf16_t)v.x; d[1] = (bf16_t)v.y; d[2] = (bf16_t)v.z; d[3] = (bf16_t)v.w;
}

// Prefetch next K-step of the same weight tile (read-once 805MB stream is the
// HBM floor; keep it ahead of the staging loads). Probe-confirmed lowering to
// global_prefetch_b8.
__device__ __forceinline__ void prefetch_w32x32(const float* __restrict__ W, int ldw,
                                                int n0, int k0, int tid) {
  __builtin_prefetch(W + (size_t)(n0 + (tid >> 3)) * ldw + k0 + (tid & 7) * 4, 0, 0);
}

// ---------------------------------------------------------------- f32 -> bf16
__global__ __launch_bounds__(256) void cvt_bf16_kernel(const float* __restrict__ x,
                                                       bf16_t* __restrict__ y, int n) {
  int i = blockIdx.x * 256 + threadIdx.x;
  if (i < n) y[i] = (bf16_t)x[i];
}

// ---------------------------------------------------------------- gating
// One wave per token: 16 dot products over D=1024, cross-lane reduce, softmax.
// Kept in exact f32: selection is discrete and precision-sensitive, cost ~134 MFLOP.
__global__ __launch_bounds__(256) void gate_kernel(const float* __restrict__ X,
                                                   const float* __restrict__ GW,
                                                   float* __restrict__ scores) {
  const int lane = threadIdx.x & 31;
  const int wv   = threadIdx.x >> 5;
  const int t    = blockIdx.x * 8 + wv;
  const float* x = X + (size_t)t * DD;

  float xv[32];
#pragma unroll
  for (int j = 0; j < 32; ++j) xv[j] = x[lane + 32 * j];

  float lg[EE];
#pragma unroll
  for (int e = 0; e < EE; ++e) {
    const float* w = GW + (size_t)e * DD;
    float s = 0.f;
#pragma unroll
    for (int j = 0; j < 32; ++j) s += xv[j] * w[lane + 32 * j];
#pragma unroll
    for (int off = 16; off > 0; off >>= 1) s += __shfl_xor(s, off, 32);
    lg[e] = s;                       // all lanes hold the full logit
  }

  float mx = lg[0];
#pragma unroll
  for (int e = 1; e < EE; ++e) mx = fmaxf(mx, lg[e]);
  float sum = 0.f;
#pragma unroll
  for (int e = 0; e < EE; ++e) { lg[e] = expf(lg[e] - mx); sum += lg[e]; }
  const float inv = 1.0f / sum;

  if (lane < EE) {
    float v = 0.f;
#pragma unroll
    for (int e = 0; e < EE; ++e) if (lane == e) v = lg[e];
    scores[(size_t)lane * TT + t] = v * inv;
  }
}

// ---------------------------------------------------------------- top-k per expert
// One 1024-thread workgroup per expert. Binary search on the uint bit pattern
// (softmax scores >= 0 -> monotone) for the C-th largest, then compact with tie fill.
__global__ __launch_bounds__(1024) void topk_kernel(const float* __restrict__ scores,
                                                    int* __restrict__ tidx,
                                                    float* __restrict__ tw) {
  __shared__ float sv[TT];
  __shared__ unsigned cnt, ngt, neq;
  const int e = blockIdx.x, tid = threadIdx.x;
  const float* row = scores + (size_t)e * TT;

  for (int i = tid; i < TT; i += 1024) sv[i] = row[i];
  __syncthreads();

  unsigned lo = 0u, hi = 0xFFFFFFFFu;   // invariant: count_ge(lo) >= C
  while (lo < hi) {
    unsigned mid = lo + ((hi - lo) >> 1) + 1u;
    if (tid == 0) cnt = 0;
    __syncthreads();
    unsigned c = 0;
    for (int i = tid; i < TT; i += 1024)
      c += (__float_as_uint(sv[i]) >= mid) ? 1u : 0u;
    if (c) atomicAdd(&cnt, c);
    __syncthreads();
    unsigned total = cnt;
    __syncthreads();                    // protect cnt before next reset
    if (total >= (unsigned)CC) lo = mid; else hi = mid - 1u;
  }
  const unsigned thr = lo;

  if (tid == 0) { ngt = 0; neq = 0; }
  __syncthreads();
  for (int i = tid; i < TT; i += 1024) {
    if (__float_as_uint(sv[i]) > thr) {
      unsigned p = atomicAdd(&ngt, 1u);
      tidx[(size_t)e * CC + p] = i;
      tw  [(size_t)e * CC + p] = sv[i];
    }
  }
  __syncthreads();
  const unsigned base = ngt;
  for (int i = tid; i < TT; i += 1024) {
    if (__float_as_uint(sv[i]) == thr) {
      unsigned p = atomicAdd(&neq, 1u);
      if (base + p < (unsigned)CC) {
        tidx[(size_t)e * CC + base + p] = i;
        tw  [(size_t)e * CC + base + p] = sv[i];
      }
    }
  }
}

// ---------------------------------------------------------------- fused up-proj
// H = gelu_exact(X@Wg^T) * (X@Wu^T), bf16 out. Block: 8 waves, tile M=512 x N=32,
// K-loop step 32. idx != null => gather token rows (expert path).
// Async path: A tiles (512x32 bf16, padded) double-buffered in LDS, copied one
// K-step ahead via GLOBAL_LOAD_ASYNC_TO_LDS_B128 (ASYNCcnt), consumed via ds_load.
// grid: (N_total/32, M_total/512, groups)
__global__ __launch_bounds__(256) void moe_up_gelu_kernel(
    const bf16_t* __restrict__ X,
    const float*  __restrict__ Wg_all,
    const float*  __restrict__ Wu_all,
    bf16_t*       __restrict__ H,
    const int*    __restrict__ idx,
    int K, int N_total, int M_total,
    unsigned long long w_stride, unsigned long long h_stride)
{
  __shared__ __align__(16) bf16_t lds_g[32 * LDSW];
  __shared__ __align__(16) bf16_t lds_u[32 * LDSW];
#if HAVE_ASYNC
  __shared__ __align__(16) bf16_t lds_a[2][512 * LDSW];
#endif

  const int tid  = threadIdx.x;
  const int lane = tid & 31;
  const int wv   = tid >> 5;
  const int hf   = lane >> 4;       // which K-half of the lane-striping
  const int l16  = lane & 15;

  const int g     = blockIdx.z;
  const int n0    = blockIdx.x * 32;
  const int m_blk = blockIdx.y * 512;

  const float* Wg = Wg_all + (size_t)g * w_stride;
  const float* Wu = Wu_all + (size_t)g * w_stride;
  bf16_t* Hg = H + (size_t)g * h_stride;

#if HAVE_ASYNC
  // copy assignment: 512 rows x 4 chunks(16B) per K-step; thread -> 8 (row,chunk)
  const bf16_t* grow[8];
  int loff[8];
#pragma unroll
  for (int j = 0; j < 8; ++j) {
    int r   = (tid >> 2) + 64 * j;
    int row = idx ? idx[(size_t)g * M_total + m_blk + r] : m_blk + r;
    grow[j] = X + (size_t)row * K + (tid & 3) * 8;
    loff[j] = r * LDSW + (tid & 3) * 8;
  }
  int afrag[4];
#pragma unroll
  for (int mt = 0; mt < 4; ++mt)
    afrag[mt] = (wv * 64 + mt * 16 + l16) * LDSW + 8 * hf;
  // prologue: prefetch first A tile
#pragma unroll
  for (int j = 0; j < 8; ++j) async_cp16(grow[j], &lds_a[0][loff[j]]);
#else
  const bf16_t* arow[4];
#pragma unroll
  for (int mt = 0; mt < 4; ++mt) {
    int m   = m_blk + wv * 64 + mt * 16 + l16;
    int row = idx ? idx[(size_t)g * M_total + m] : m;
    arow[mt] = X + (size_t)row * K + 8 * hf;
  }
#endif

  v8f ag[4][2] = {};
  v8f au[4][2] = {};

  for (int k0 = 0; k0 < K; k0 += 32) {
#if HAVE_ASYNC
    const int cur = (k0 >> 5) & 1;
    wait_async0();                 // this wave's tile-k copies have landed
#endif
    __syncthreads();               // all waves' copies visible; prev-buffer reads done
    stage_w32x32(Wg, K, n0, k0, lds_g, tid);
    stage_w32x32(Wu, K, n0, k0, lds_u, tid);
    if (k0 + 32 < K) {
      prefetch_w32x32(Wg, K, n0, k0 + 32, tid);
      prefetch_w32x32(Wu, K, n0, k0 + 32, tid);
#if HAVE_ASYNC
#pragma unroll
      for (int j = 0; j < 8; ++j)
        async_cp16(grow[j] + k0 + 32, &lds_a[1 - cur][loff[j]]);
#endif
    }
    __syncthreads();

    v16bf a[4];
#pragma unroll
    for (int mt = 0; mt < 4; ++mt)
#if HAVE_ASYNC
      a[mt] = load_tile_bf16(&lds_a[cur][afrag[mt]]);
#else
      a[mt] = load_tile_bf16(arow[mt] + k0);
#endif

    v16bf bg[2], bu[2];
#pragma unroll
    for (int nt = 0; nt < 2; ++nt) {
      bg[nt] = load_tile_bf16(lds_g + (nt * 16 + l16) * LDSW + 8 * hf);
      bu[nt] = load_tile_bf16(lds_u + (nt * 16 + l16) * LDSW + 8 * hf);
    }

#pragma unroll
    for (int mt = 0; mt < 4; ++mt)
#pragma unroll
      for (int nt = 0; nt < 2; ++nt) {
        ag[mt][nt] = wmma_bf16(a[mt], bg[nt], ag[mt][nt]);
        au[mt][nt] = wmma_bf16(a[mt], bu[nt], au[mt][nt]);
      }
  }

#pragma unroll
  for (int mt = 0; mt < 4; ++mt)
#pragma unroll
    for (int nt = 0; nt < 2; ++nt)
#pragma unroll
      for (int r = 0; r < 8; ++r) {
        int m = m_blk + wv * 64 + mt * 16 + hf * 8 + r;
        int n = n0 + nt * 16 + l16;
        float gv = ag[mt][nt][r];
        float uv = au[mt][nt][r];
        float hv = 0.5f * gv * (1.0f + erff(gv * 0.70710678118654752f)) * uv;
        Hg[(size_t)m * N_total + n] = (bf16_t)hv;
      }
}

// ---------------------------------------------------------------- down-proj
// Out += / = (H @ Wd^T) [* tw, scattered by idx]. Same tiling/pipeline as up.
// grid: (N_out/32, M_total/512, groups)
__global__ __launch_bounds__(256) void moe_down_kernel(
    const bf16_t* __restrict__ Hsrc,
    const float*  __restrict__ Wd_all,
    float*        __restrict__ Out,
    const int*    __restrict__ idx,
    const float*  __restrict__ tw,
    int K, int N_out, int M_total,
    unsigned long long w_stride, unsigned long long h_stride)
{
  __shared__ __align__(16) bf16_t lds_w[32 * LDSW];
#if HAVE_ASYNC
  __shared__ __align__(16) bf16_t lds_a[2][512 * LDSW];
#endif
  __shared__ int   sidx[512];
  __shared__ float swt[512];

  const int tid  = threadIdx.x;
  const int lane = tid & 31;
  const int wv   = tid >> 5;
  const int hf   = lane >> 4;
  const int l16  = lane & 15;

  const int g     = blockIdx.z;
  const int n0    = blockIdx.x * 32;
  const int m_blk = blockIdx.y * 512;

  const float*  Wd = Wd_all + (size_t)g * w_stride;
  const bf16_t* Hg = Hsrc  + (size_t)g * h_stride;

  if (idx) {
    for (int j = tid; j < 512; j += 256) {
      sidx[j] = idx[(size_t)g * M_total + m_blk + j];
      swt[j]  = tw [(size_t)g * M_total + m_blk + j];
    }
  }

#if HAVE_ASYNC
  const bf16_t* grow[8];
  int loff[8];
#pragma unroll
  for (int j = 0; j < 8; ++j) {
    int r   = (tid >> 2) + 64 * j;
    grow[j] = Hg + (size_t)(m_blk + r) * K + (tid & 3) * 8;
    loff[j] = r * LDSW + (tid & 3) * 8;
  }
  int afrag[4];
#pragma unroll
  for (int mt = 0; mt < 4; ++mt)
    afrag[mt] = (wv * 64 + mt * 16 + l16) * LDSW + 8 * hf;
#pragma unroll
  for (int j = 0; j < 8; ++j) async_cp16(grow[j], &lds_a[0][loff[j]]);
#else
  const bf16_t* arow[4];
#pragma unroll
  for (int mt = 0; mt < 4; ++mt) {
    int m = m_blk + wv * 64 + mt * 16 + l16;
    arow[mt] = Hg + (size_t)m * K + 8 * hf;
  }
#endif

  v8f acc[4][2] = {};

  for (int k0 = 0; k0 < K; k0 += 32) {
#if HAVE_ASYNC
    const int cur = (k0 >> 5) & 1;
    wait_async0();
#endif
    __syncthreads();
    stage_w32x32(Wd, K, n0, k0, lds_w, tid);
    if (k0 + 32 < K) {
      prefetch_w32x32(Wd, K, n0, k0 + 32, tid);
#if HAVE_ASYNC
#pragma unroll
      for (int j = 0; j < 8; ++j)
        async_cp16(grow[j] + k0 + 32, &lds_a[1 - cur][loff[j]]);
#endif
    }
    __syncthreads();

    v16bf a[4];
#pragma unroll
    for (int mt = 0; mt < 4; ++mt)
#if HAVE_ASYNC
      a[mt] = load_tile_bf16(&lds_a[cur][afrag[mt]]);
#else
      a[mt] = load_tile_bf16(arow[mt] + k0);
#endif

    v16bf b[2];
#pragma unroll
    for (int nt = 0; nt < 2; ++nt)
      b[nt] = load_tile_bf16(lds_w + (nt * 16 + l16) * LDSW + 8 * hf);

#pragma unroll
    for (int mt = 0; mt < 4; ++mt)
#pragma unroll
      for (int nt = 0; nt < 2; ++nt)
        acc[mt][nt] = wmma_bf16(a[mt], b[nt], acc[mt][nt]);
  }

  if (idx) {
#pragma unroll
    for (int mt = 0; mt < 4; ++mt)
#pragma unroll
      for (int nt = 0; nt < 2; ++nt)
#pragma unroll
        for (int r = 0; r < 8; ++r) {
          int ml = wv * 64 + mt * 16 + hf * 8 + r;       // 0..511 in block
          int n  = n0 + nt * 16 + l16;
          float v = acc[mt][nt][r] * swt[ml];
          atomicAdd(&Out[(size_t)sidx[ml] * N_out + n], v);
        }
  } else {
#pragma unroll
    for (int mt = 0; mt < 4; ++mt)
#pragma unroll
      for (int nt = 0; nt < 2; ++nt)
#pragma unroll
        for (int r = 0; r < 8; ++r) {
          int m = m_blk + wv * 64 + mt * 16 + hf * 8 + r;
          int n = n0 + nt * 16 + l16;
          Out[(size_t)m * N_out + n] = acc[mt][nt][r];
        }
  }
}

// ---------------------------------------------------------------- launch
extern "C" void kernel_launch(void* const* d_in, const int* in_sizes, int n_in,
                              void* d_out, int out_size, void* d_ws, size_t ws_size,
                              hipStream_t stream) {
  const float* x  = (const float*)d_in[0];   // (B,S,D)
  const float* gw = (const float*)d_in[1];   // (E,D)
  const float* Wg = (const float*)d_in[2];   // (E,I,D)
  const float* Wu = (const float*)d_in[3];   // (E,I,D)
  const float* Wd = (const float*)d_in[4];   // (E,D,I)
  const float* Sg = (const float*)d_in[5];   // (IS,D)
  const float* Su = (const float*)d_in[6];   // (IS,D)
  const float* Sd = (const float*)d_in[7];   // (D,IS)
  float* out = (float*)d_out;                // (B,S,D)

  char* ws = (char*)d_ws;
  size_t off = 0;
  bf16_t* xbf    = (bf16_t*)(ws + off); off += (size_t)TT * DD * 2;        //  8 MB
  float*  scores = (float*) (ws + off); off += (size_t)EE * TT * 4;        // 256 KB
  int*    tidx   = (int*)   (ws + off); off += (size_t)EE * CC * 4;        //  32 KB
  float*  tw     = (float*) (ws + off); off += (size_t)EE * CC * 4;        //  32 KB
  bf16_t* hsh    = (bf16_t*)(ws + off); off += (size_t)TT * ISH * 2;       //  16 MB
  bf16_t* hex    = (bf16_t*)(ws + off); off += (size_t)EE * CC * II * 2;   //  64 MB
  (void)off; (void)ws_size; (void)in_sizes; (void)n_in; (void)out_size;

  // 1) activations f32 -> bf16 (serves gathered-expert and shared-expert paths)
  cvt_bf16_kernel<<<(TT * DD) / 256, 256, 0, stream>>>(x, xbf, TT * DD);

  // 2) gating logits + softmax -> (E,T) scores, exact f32
  gate_kernel<<<TT / 8, 256, 0, stream>>>(x, gw, scores);

  // 3) expert-choice top-C per expert
  topk_kernel<<<EE, 1024, 0, stream>>>(scores, tidx, tw);

  // 4) shared-expert up: h = gelu(X Sg^T) * (X Su^T), (T, IS)
  moe_up_gelu_kernel<<<dim3(ISH / 32, TT / 512, 1), 256, 0, stream>>>(
      xbf, Sg, Su, hsh, nullptr, DD, ISH, TT, 0ull, 0ull);

  // 5) expert up (gathered): (E, C, I)
  moe_up_gelu_kernel<<<dim3(II / 32, 1, EE), 256, 0, stream>>>(
      xbf, Wg, Wu, hex, tidx, DD, II, CC,
      (unsigned long long)II * DD, (unsigned long long)CC * II);

  // 6) shared-expert down: out = h_sh Sd^T (direct store, initializes out)
  moe_down_kernel<<<dim3(DD / 32, TT / 512, 1), 256, 0, stream>>>(
      hsh, Sd, out, nullptr, nullptr, ISH, DD, TT, 0ull, 0ull);

  // 7) expert down: out += tw * (h_e Wd^T) scattered by token index (atomic f32)
  moe_down_kernel<<<dim3(DD / 32, 1, EE), 256, 0, stream>>>(
      hex, Wd, out, tidx, tw, II, DD, CC,
      (unsigned long long)DD * II, (unsigned long long)CC * II);
}
